// pendulumRNNMHA_20538533609628
// MI455X (gfx1250) — matrix operand order repaired and checked by reading, not compiled, not run on hardware
//
#include <hip/hip_runtime.h>
#include <hip/hip_bf16.h>
#include <math.h>

// ---------------------------------------------------------------------------
// 3-layer LSTM (B=256, T=512, H=128, FUT=512) persistent WMMA kernel, gfx1250
// ---------------------------------------------------------------------------

typedef __attribute__((ext_vector_type(16))) _Float16 v16h;
typedef __attribute__((ext_vector_type(8)))  _Float16 v8h;
typedef __attribute__((ext_vector_type(8)))  float    v8f;

#define HN   128          // hidden size
#define GN   512          // 4*H gate rows
#define TT   1024         // T + FUT
#define TIN  512          // teacher-forced steps
#define HP   136          // padded LDS row stride (halves), 272B = 16B-aligned
#define SH_TILE (16*HP)   // one 16xH (padded) f16 tile in LDS

// f16 weight region offsets (halves) inside d_ws
#define O_WHH1 0
#define O_WIH2 65536
#define O_WHH2 131072
#define O_WIH3 196608
#define O_WHH3 262144
#define O_WV   327680
#define O_OW   344064
#define N_HALF 360448               // total halves
#define F_BYTE_OFF 720896           // f32 section byte offset (=N_HALF*2)

union HalfVec { v16h v; v8h h[2]; };

__device__ inline v8f vzero() {
  v8f z = {0.f,0.f,0.f,0.f,0.f,0.f,0.f,0.f};
  return z;
}

__device__ inline float fsig(float x) { return 1.0f / (1.0f + __expf(-x)); }

// A fragment (16x32 f16) from LDS h-buffer, row stride HP.
// base = hbuf + k0 ; doc layout: lane<16 holds K0-7,K16-23; lane>=16 K8-15,K24-31
__device__ inline v16h load_a_lds(const _Float16* base, int lane) {
  const int r    = lane & 15;
  const int koff = (lane & 16) ? 8 : 0;
  const _Float16* p = base + r * HP + koff;
  HalfVec u;
  u.h[0] = *(const v8h*)(p);
  u.h[1] = *(const v8h*)(p + 16);
  return u.v;
}

// B fragment (32x16 f16) = W^T tile, W row-major (rows=out col, K contiguous).
// wrow = W + (n_row)*HN + k0 ; lane<16: K 0-7,16-23 ; lane>=16: K 8-15,24-31
__device__ inline v16h load_b_gl(const _Float16* wrow, int lane) {
  const _Float16* p = wrow + ((lane & 16) ? 8 : 0);
  HalfVec u;
  u.h[0] = *(const v8h*)(p);
  u.h[1] = *(const v8h*)(p + 16);
  return u.v;
}

__device__ inline v8f wmma16(v16h a, v16h b, v8f c) {
  return __builtin_amdgcn_wmma_f32_16x16x32_f16(
      /*neg_a=*/false, a, /*neg_b=*/false, b,
      /*c_mod=*/(short)0, c, /*reuse_a=*/false, /*reuse_b=*/false);
}

// --------------------------- prep kernels ----------------------------------

__global__ void prepW(const float* __restrict__ Whh1, const float* __restrict__ Wih2,
                      const float* __restrict__ Whh2, const float* __restrict__ Wih3,
                      const float* __restrict__ Whh3, const float* __restrict__ in_proj_w,
                      const float* __restrict__ out_w, _Float16* __restrict__ WHd) {
  int i = blockIdx.x * blockDim.x + threadIdx.x;
  if (i >= N_HALF) return;
  float v;
  if      (i < O_WIH2) v = Whh1[i];
  else if (i < O_WHH2) v = Wih2[i - O_WIH2];
  else if (i < O_WIH3) v = Whh2[i - O_WHH2];
  else if (i < O_WHH3) v = Wih3[i - O_WIH3];
  else if (i < O_WV)   v = Whh3[i - O_WHH3];
  else if (i < O_OW)   v = in_proj_w[2 * HN * HN + (i - O_WV)];  // Wv = rows 2H..3H
  else                 v = out_w[i - O_OW];
  WHd[i] = (_Float16)v;
}

// f32 side data: fused biases, bv, out_b, lin_w, Wih1 column, lin_b
__global__ void prepF(const float* __restrict__ bih1, const float* __restrict__ bhh1,
                      const float* __restrict__ bih2, const float* __restrict__ bhh2,
                      const float* __restrict__ bih3, const float* __restrict__ bhh3,
                      const float* __restrict__ in_proj_b, const float* __restrict__ out_b,
                      const float* __restrict__ lin_w, const float* __restrict__ Wih1,
                      const float* __restrict__ lin_b, float* __restrict__ F) {
  int i = blockIdx.x * blockDim.x + threadIdx.x;
  if      (i < 512)  F[i] = bih1[i] + bhh1[i];
  else if (i < 1024) F[i] = bih2[i - 512] + bhh2[i - 512];
  else if (i < 1536) F[i] = bih3[i - 1024] + bhh3[i - 1024];
  else if (i < 1664) F[i] = in_proj_b[256 + (i - 1536)];   // bv
  else if (i < 1792) F[i] = out_b[i - 1664];
  else if (i < 1920) F[i] = lin_w[i - 1792];
  else if (i < 2432) F[i] = Wih1[i - 1920];
  else if (i == 2432) F[i] = lin_b[0];
}

// --------------------------- main persistent kernel ------------------------

__global__ __launch_bounds__(256) void lstm_main(const float* __restrict__ input,
                                                 const _Float16* __restrict__ WH,
                                                 const float* __restrict__ F,
                                                 float* __restrict__ out) {
  // LDS: 6 double-buffered h tiles + attn1 tile (f16), x feed + out accum (f32)
  __shared__ _Float16 sh[7 * SH_TILE];
  __shared__ float sx[16];
  __shared__ float sacc[16];

  _Float16* h1a = sh + 0 * SH_TILE;
  _Float16* h1b = sh + 1 * SH_TILE;
  _Float16* h2a = sh + 2 * SH_TILE;
  _Float16* h2b = sh + 3 * SH_TILE;
  _Float16* h3a = sh + 4 * SH_TILE;
  _Float16* h3b = sh + 5 * SH_TILE;
  _Float16* at1 = sh + 6 * SH_TILE;

  const int tid  = threadIdx.x;
  const int lane = tid & 31;
  const int wid  = tid >> 5;          // 8 waves
  const int n0   = wid * 16;          // hidden-column slice owned by this wave
  const int ln16 = lane & 15;
  const int hib  = (lane & 16) ? 8 : 0;
  const int col  = n0 + ln16;         // hidden column handled by this lane
  const int m0   = blockIdx.x * 16;   // batch tile base row

  // zero initial h state (both buffers) + accumulators
  for (int i = tid; i < 7 * SH_TILE; i += 256) sh[i] = (_Float16)0.0f;
  if (tid < 16) { sacc[tid] = 0.0f; sx[tid] = 0.0f; }
  __syncthreads();

  // per-lane constant parameters (column 'col' of each bias / vector)
  const float b1i = F[col],        b1f = F[128 + col],  b1g = F[256 + col],  b1o = F[384 + col];
  const float b2i = F[512 + col],  b2f = F[640 + col],  b2g = F[768 + col],  b2o = F[896 + col];
  const float b3i = F[1024 + col], b3f = F[1152 + col], b3g = F[1280 + col], b3o = F[1408 + col];
  const float bvv = F[1536 + col];
  const float obb = F[1664 + col];
  const float lw  = F[1792 + col];
  const float w1i = F[1920 + col], w1f = F[2048 + col], w1g = F[2176 + col], w1o = F[2304 + col];
  const float linb = F[2432];

  // cell state tiles live in registers, exactly in WMMA C/D layout
  v8f c1 = vzero(), c2 = vzero(), c3 = vzero();

  for (int t = 0; t < TT; ++t) {
    const _Float16* h1r = (t & 1) ? h1b : h1a;  _Float16* h1w = (t & 1) ? h1a : h1b;
    const _Float16* h2r = (t & 1) ? h2b : h2a;  _Float16* h2w = (t & 1) ? h2a : h2b;
    const _Float16* h3r = (t & 1) ? h3b : h3a;  _Float16* h3w = (t & 1) ? h3a : h3b;

    // Make weight addresses loop-variant via an opaque ZERO offset so LLVM
    // cannot hoist all B fragments out of the t-loop, while keeping the
    // pointer provenance (kernel arg -> addrspace(1)) intact so the loads
    // lower to global_load_b128 (LOADcnt only), not flat_load (LOADcnt+DScnt).
    int woff = 0;
    asm volatile("" : "+s"(woff));
    const _Float16* WB   = WH + woff;
    const _Float16* WHH1 = WB + O_WHH1;
    const _Float16* WIH2 = WB + O_WIH2;
    const _Float16* WHH2 = WB + O_WHH2;
    const _Float16* WIH3 = WB + O_WIH3;
    const _Float16* WHH3 = WB + O_WHH3;
    const _Float16* WV   = WB + O_WV;
    const _Float16* OW   = WB + O_OW;

    // fetch this step's scalar input (teacher-forced phase)
    if (t < TIN && wid == 0 && lane < 16)
      sx[lane] = input[(m0 + lane) * TIN + t];
    __syncthreads();

    float xv[8];
    #pragma unroll
    for (int v = 0; v < 8; ++v) xv[v] = sx[v + hib];

    // ---------------- LSTM layer 1: gates = h1_old @ Whh1^T (+x*Wih1 +b) ---
    v8f g0 = vzero(), g1 = vzero(), g2 = vzero(), g3 = vzero();
    #pragma unroll
    for (int kt = 0; kt < 4; ++kt) {
      v16h a = load_a_lds(h1r + kt * 32, lane);
      const _Float16* wr = WHH1 + col * HN + kt * 32;
      g0 = wmma16(a, load_b_gl(wr,                 lane), g0);
      g1 = wmma16(a, load_b_gl(wr + 1 * HN * HN,   lane), g1);
      g2 = wmma16(a, load_b_gl(wr + 2 * HN * HN,   lane), g2);
      g3 = wmma16(a, load_b_gl(wr + 3 * HN * HN,   lane), g3);
    }
    #pragma unroll
    for (int v = 0; v < 8; ++v) {
      float ig = fsig (g0[v] + b1i + xv[v] * w1i);
      float fg = fsig (g1[v] + b1f + xv[v] * w1f);
      float gg = tanhf(g2[v] + b1g + xv[v] * w1g);
      float og = fsig (g3[v] + b1o + xv[v] * w1o);
      float cn = fg * c1[v] + ig * gg;
      c1[v] = cn;
      h1w[(v + hib) * HP + col] = (_Float16)(og * tanhf(cn));
    }
    __syncthreads();

    // ---------------- LSTM layer 2: h1_new@Wih2^T + h2_old@Whh2^T + b ------
    g0 = vzero(); g1 = vzero(); g2 = vzero(); g3 = vzero();
    #pragma unroll
    for (int kt = 0; kt < 4; ++kt) {
      v16h a = load_a_lds(h1w + kt * 32, lane);
      const _Float16* wr = WIH2 + col * HN + kt * 32;
      g0 = wmma16(a, load_b_gl(wr,               lane), g0);
      g1 = wmma16(a, load_b_gl(wr + 1 * HN * HN, lane), g1);
      g2 = wmma16(a, load_b_gl(wr + 2 * HN * HN, lane), g2);
      g3 = wmma16(a, load_b_gl(wr + 3 * HN * HN, lane), g3);
    }
    #pragma unroll
    for (int kt = 0; kt < 4; ++kt) {
      v16h a = load_a_lds(h2r + kt * 32, lane);
      const _Float16* wr = WHH2 + col * HN + kt * 32;
      g0 = wmma16(a, load_b_gl(wr,               lane), g0);
      g1 = wmma16(a, load_b_gl(wr + 1 * HN * HN, lane), g1);
      g2 = wmma16(a, load_b_gl(wr + 2 * HN * HN, lane), g2);
      g3 = wmma16(a, load_b_gl(wr + 3 * HN * HN, lane), g3);
    }
    #pragma unroll
    for (int v = 0; v < 8; ++v) {
      float ig = fsig (g0[v] + b2i);
      float fg = fsig (g1[v] + b2f);
      float gg = tanhf(g2[v] + b2g);
      float og = fsig (g3[v] + b2o);
      float cn = fg * c2[v] + ig * gg;
      c2[v] = cn;
      h2w[(v + hib) * HP + col] = (_Float16)(og * tanhf(cn));
    }
    __syncthreads();

    // ---------------- LSTM layer 3: h2_new@Wih3^T + h3_old@Whh3^T + b ------
    g0 = vzero(); g1 = vzero(); g2 = vzero(); g3 = vzero();
    #pragma unroll
    for (int kt = 0; kt < 4; ++kt) {
      v16h a = load_a_lds(h2w + kt * 32, lane);
      const _Float16* wr = WIH3 + col * HN + kt * 32;
      g0 = wmma16(a, load_b_gl(wr,               lane), g0);
      g1 = wmma16(a, load_b_gl(wr + 1 * HN * HN, lane), g1);
      g2 = wmma16(a, load_b_gl(wr + 2 * HN * HN, lane), g2);
      g3 = wmma16(a, load_b_gl(wr + 3 * HN * HN, lane), g3);
    }
    #pragma unroll
    for (int kt = 0; kt < 4; ++kt) {
      v16h a = load_a_lds(h3r + kt * 32, lane);
      const _Float16* wr = WHH3 + col * HN + kt * 32;
      g0 = wmma16(a, load_b_gl(wr,               lane), g0);
      g1 = wmma16(a, load_b_gl(wr + 1 * HN * HN, lane), g1);
      g2 = wmma16(a, load_b_gl(wr + 2 * HN * HN, lane), g2);
      g3 = wmma16(a, load_b_gl(wr + 3 * HN * HN, lane), g3);
    }
    #pragma unroll
    for (int v = 0; v < 8; ++v) {
      float ig = fsig (g0[v] + b3i);
      float fg = fsig (g1[v] + b3f);
      float gg = tanhf(g2[v] + b3g);
      float og = fsig (g3[v] + b3o);
      float cn = fg * c3[v] + ig * gg;
      c3[v] = cn;
      h3w[(v + hib) * HP + col] = (_Float16)(og * tanhf(cn));
    }
    __syncthreads();

    // ---------------- head stage 1: attn1 = h3_new @ Wv^T + bv -------------
    v8f va = vzero();
    #pragma unroll
    for (int kt = 0; kt < 4; ++kt) {
      v16h a = load_a_lds(h3w + kt * 32, lane);
      va = wmma16(a, load_b_gl(WV + col * HN + kt * 32, lane), va);
    }
    #pragma unroll
    for (int v = 0; v < 8; ++v)
      at1[(v + hib) * HP + col] = (_Float16)(va[v] + bvv);
    __syncthreads();

    // ---------------- head stage 2: attn2 = attn1 @ out_w^T + out_b --------
    v8f vb = vzero();
    #pragma unroll
    for (int kt = 0; kt < 4; ++kt) {
      v16h a = load_a_lds(at1 + kt * 32, lane);
      vb = wmma16(a, load_b_gl(OW + col * HN + kt * 32, lane), vb);
    }

    // ---------------- final dot with lin_w: shuffle-reduce + LDS atomics ---
    #pragma unroll
    for (int v = 0; v < 8; ++v) {
      float p = (vb[v] + obb) * lw;
      p += __shfl_xor(p, 1, 16);
      p += __shfl_xor(p, 2, 16);
      p += __shfl_xor(p, 4, 16);
      p += __shfl_xor(p, 8, 16);
      if (ln16 == 0) atomicAdd(&sacc[v + hib], p);   // ds_add_f32
    }
    __syncthreads();

    if (wid == 0 && lane < 16) {
      float o = sacc[lane] + linb;
      out[(m0 + lane) * TT + t] = o;   // output, (B, T+FUT) row-major
      sx[lane]   = o;                  // autoregressive feedback
      sacc[lane] = 0.0f;
    }
    __syncthreads();
  }
}

// --------------------------- launcher --------------------------------------

extern "C" void kernel_launch(void* const* d_in, const int* in_sizes, int n_in,
                              void* d_out, int out_size, void* d_ws, size_t ws_size,
                              hipStream_t stream) {
  const float* input     = (const float*)d_in[0];
  const float* Wih1      = (const float*)d_in[2];
  const float* Whh1      = (const float*)d_in[3];
  const float* bih1      = (const float*)d_in[4];
  const float* bhh1      = (const float*)d_in[5];
  const float* Wih2      = (const float*)d_in[6];
  const float* Whh2      = (const float*)d_in[7];
  const float* bih2      = (const float*)d_in[8];
  const float* bhh2      = (const float*)d_in[9];
  const float* Wih3      = (const float*)d_in[10];
  const float* Whh3      = (const float*)d_in[11];
  const float* bih3      = (const float*)d_in[12];
  const float* bhh3      = (const float*)d_in[13];
  const float* in_proj_w = (const float*)d_in[14];
  const float* in_proj_b = (const float*)d_in[15];
  const float* out_w     = (const float*)d_in[16];
  const float* out_b     = (const float*)d_in[17];
  const float* lin_w     = (const float*)d_in[18];
  const float* lin_b     = (const float*)d_in[19];

  _Float16* WHd = (_Float16*)d_ws;
  float*    F   = (float*)((char*)d_ws + F_BYTE_OFF);

  prepW<<<(N_HALF + 255) / 256, 256, 0, stream>>>(Whh1, Wih2, Whh2, Wih3, Whh3,
                                                  in_proj_w, out_w, WHd);
  prepF<<<10, 256, 0, stream>>>(bih1, bhh1, bih2, bhh2, bih3, bhh3,
                                in_proj_b, out_b, lin_w, Wih1, lin_b, F);
  lstm_main<<<16, 256, 0, stream>>>(input, WHd, F, (float*)d_out);
}
